// SimpleESN_72782515798999
// MI455X (gfx1250) — matrix operand order repaired and checked by reading, not compiled, not run on hardware
//
#include <hip/hip_runtime.h>
#include <hip/hip_bf16.h>

// ---------------------------------------------------------------------------
// Persistent-kernel leaky ESN for gfx1250 (MI455X).
//   h_t = 0.7*h_{t-1} + 0.3*tanh([W | W_in] @ [h_{t-1}; x_t])
// Each WG pins its 16 rows of [W|W_in] (f16, zero-padded to K=4352) in WGP LDS
// once, then runs all 2048 steps with v_wmma_f32_16x16x32_f16 matvecs
// (h in A row 0, W rows in B) and a device-wide atomic barrier between steps.
// A-fragment inactive lanes read a dedicated zeroed LDS strip so the k-loop is
// a branch-free, EXEC-stable stream of ds_load_b128 + wmma. x_{t+1} is
// register-pipelined across steps (issued after the phase-1 barrier so its
// latency hides behind the WMMA phase); two accumulator chains break the
// WMMA D->C dependency chain.
// ---------------------------------------------------------------------------

typedef _Float16 half4  __attribute__((ext_vector_type(4)));
typedef _Float16 half8  __attribute__((ext_vector_type(8)));
typedef _Float16 half16 __attribute__((ext_vector_type(16)));
typedef float    f32x8  __attribute__((ext_vector_type(8)));

#define T_STEPS   2048
#define RES       4096
#define IN_DIM    128
#define KTOT      (RES + IN_DIM)        // 4224 live reduction length
#define KPAD      4352                  // padded to 136 k-tiles = 8 waves * 17
#define ROWS_WG   16                    // output rows owned per WG (one N-tile)
#define NWG       (RES / ROWS_WG)       // 256 workgroups
#define BLOCK     256                   // 8 waves (wave32)
#define NWAVES    (BLOCK / 32)
#define TPW       ((KPAD / 32) / NWAVES) // 17 k-tiles per wave
#define LSTRIDE   (KPAD + 8)            // 4360 halves; +8 staggers LDS banks
#define ZLEN      (TPW * 32)            // 544-half zero strip for inactive lanes
#define LEAK      0.3f

__global__ __launch_bounds__(BLOCK, 1)
void esn_persistent_kernel(const float* __restrict__ x,
                           const float* __restrict__ Win,
                           const float* __restrict__ W,
                           float* __restrict__ out,
                           _Float16* __restrict__ hbufA,
                           _Float16* __restrict__ hbufB,
                           unsigned int* __restrict__ bar) {
  // 16*4360*2 = 139,520 B W  + 8,704 B hx + 1,088 B zeros + scratch  (<160KB)
  __shared__ __align__(16) _Float16 Wl[ROWS_WG * LSTRIDE];
  __shared__ __align__(16) _Float16 hx[KPAD];
  __shared__ __align__(16) _Float16 zbuf[ZLEN];
  __shared__ float red[NWAVES][ROWS_WG];
  __shared__ float hstate[ROWS_WG];

  const unsigned tid  = threadIdx.x;
  const unsigned lane = tid & 31u;
  const unsigned wave = tid >> 5;
  const unsigned row0 = blockIdx.x * ROWS_WG;

  // ---- one-time: stage this WG's rows of [W | W_in] into LDS as f16 -------
  for (unsigned r = wave; r < ROWS_WG; r += NWAVES) {
    const float4* srcW = (const float4*)(W + (size_t)(row0 + r) * RES);
    _Float16* dst = &Wl[r * LSTRIDE];
    for (unsigned j = lane; j < RES / 4; j += 32u) {
      float4 v = srcW[j];
      half4 h = {(_Float16)v.x, (_Float16)v.y, (_Float16)v.z, (_Float16)v.w};
      *(half4*)&dst[4u * j] = h;
    }
    const float4* srcI = (const float4*)(Win + (size_t)(row0 + r) * IN_DIM);
    if (lane < IN_DIM / 4) {
      float4 v = srcI[lane];
      half4 h = {(_Float16)v.x, (_Float16)v.y, (_Float16)v.z, (_Float16)v.w};
      *(half4*)&dst[RES + 4u * lane] = h;
    }
  }
  // zero the K pad of each W row, the hx pad, and the zero strip
  for (unsigned p = tid; p < ROWS_WG * (LSTRIDE - KTOT); p += BLOCK) {
    const unsigned r = p / (LSTRIDE - KTOT);
    const unsigned c = p % (LSTRIDE - KTOT);
    Wl[r * LSTRIDE + KTOT + c] = (_Float16)0.0f;
  }
  if (tid < KPAD - KTOT) hx[KTOT + tid] = (_Float16)0.0f;
  for (unsigned p = tid; p < ZLEN; p += BLOCK) zbuf[p] = (_Float16)0.0f;
  if (tid < ROWS_WG) hstate[tid] = 0.0f;
  __syncthreads();

  // B operand: lane l supplies W row (row0 + (l&15)), K-half (l>>4)*16.
  const unsigned rsel = lane >> 4;                        // 0 or 1
  const unsigned kb0  = wave * TPW * 32u;                 // this wave's K base
  const half8* bbase = (const half8*)&Wl[(lane & 15u) * LSTRIDE + rsel * 16u + kb0];
  // A operand: only lanes 0 and 16 carry hx data; others read the zero strip.
  const bool act = (lane & 15u) == 0u;
  const _Float16* abase = act ? &hx[kb0 + rsel * 16u] : &zbuf[0];

  // register-pipelined input vector: xreg holds x_t during step t
  float xreg = (tid < IN_DIM) ? x[tid] : 0.0f;

  for (unsigned t = 0; t < T_STEPS; ++t) {
    const _Float16* hprev = (t & 1u) ? hbufB : hbufA;
    _Float16*       hcur  = (t & 1u) ? hbufA : hbufB;

    // ---- phase 1: gather h_{t-1} (f16) and x_t into LDS hx ---------------
    {
      const uint4* src = (const uint4*)hprev;             // 4096 halves = 512 uint4
      uint4* dst = (uint4*)hx;
      uint4 h0 = src[tid];
      uint4 h1 = src[tid + BLOCK];
      dst[tid] = h0;
      dst[tid + BLOCK] = h1;
      if (tid < IN_DIM) hx[RES + tid] = (_Float16)xreg;
    }
    __syncthreads();

    // issue next step's x load now (post-barrier): consumed only by next
    // iteration's ds_store, so WMMA phase + grid barrier hide its latency
    float xnext = 0.0f;
    if (tid < IN_DIM && (t + 1u) < T_STEPS)
      xnext = x[(size_t)(t + 1u) * IN_DIM + tid];

    // ---- phase 2: branch-free WMMA matvec, two accumulator chains --------
    f32x8 c0 = {};
    f32x8 c1 = {};
#pragma unroll
    for (unsigned kt = 0; kt < TPW; ++kt) {
      // B fragment: two contiguous ds_load_b128 of this lane's W row slice
      half8 blo = bbase[2u * kt];
      half8 bhi = bbase[2u * kt + 1u];
      half16 b = __builtin_shufflevector(blo, bhi, 0, 1, 2, 3, 4, 5, 6, 7, 8,
                                         9, 10, 11, 12, 13, 14, 15);
      // A fragment: hx (lanes 0/16) or the persistent zero strip (others)
      half8 alo = *(const half8*)&abase[kt * 32u];
      half8 ahi = *(const half8*)&abase[kt * 32u + 8u];
      half16 a = __builtin_shufflevector(alo, ahi, 0, 1, 2, 3, 4, 5, 6, 7, 8,
                                         9, 10, 11, 12, 13, 14, 15);
      if (kt & 1u)
        c1 = __builtin_amdgcn_wmma_f32_16x16x32_f16(false, a, false, b,
                                                    (short)0, c1, false, false);
      else
        c0 = __builtin_amdgcn_wmma_f32_16x16x32_f16(false, a, false, b,
                                                    (short)0, c0, false, false);
    }
    // D[0][n] lives in VGPR0 of lanes 0..15; merge the two chains here
    if (lane < 16u) red[wave][lane] = c0[0] + c1[0];
    __syncthreads();

    // ---- phase 3: cross-wave reduce, leak + tanh, publish h_t ------------
    if (tid < ROWS_WG) {
      float y = 0.0f;
#pragma unroll
      for (unsigned w = 0; w < NWAVES; ++w) y += red[w][tid];
      float hn = (1.0f - LEAK) * hstate[tid] + LEAK * tanhf(y);
      hstate[tid] = hn;
      out[(size_t)t * RES + row0 + tid] = hn;
      hcur[row0 + tid] = (_Float16)hn;
    }

    // ---- phase 4: device-wide barrier (monotonic counter) ----------------
    __threadfence();
    __syncthreads();
    if (tid == 0u) {
      __hip_atomic_fetch_add(bar, 1u, __ATOMIC_ACQ_REL,
                             __HIP_MEMORY_SCOPE_AGENT);
      const unsigned target = (t + 1u) * (unsigned)NWG;
      while (__hip_atomic_load(bar, __ATOMIC_ACQUIRE,
                               __HIP_MEMORY_SCOPE_AGENT) < target) {
        __builtin_amdgcn_s_sleep(1);
      }
    }
    __syncthreads();
    xreg = xnext;
  }
}

// Workspace layout: [0,4): barrier counter | [256, 256+8K): hbufA | +8K: hbufB
#define WS_BAR_OFF 0
#define WS_HA_OFF  256
#define WS_HB_OFF  (256 + RES * 2)
#define WS_INIT    (256 + 2 * RES * 2)

extern "C" void kernel_launch(void* const* d_in, const int* in_sizes, int n_in,
                              void* d_out, int out_size, void* d_ws,
                              size_t ws_size, hipStream_t stream) {
  (void)in_sizes; (void)n_in; (void)out_size; (void)ws_size;
  const float* x   = (const float*)d_in[0];   // [T, 128]
  const float* Win = (const float*)d_in[1];   // [4096, 128]
  const float* W   = (const float*)d_in[2];   // [4096, 4096]
  float* out = (float*)d_out;                 // [T, 4096]

  char* ws = (char*)d_ws;
  unsigned int* bar = (unsigned int*)(ws + WS_BAR_OFF);
  _Float16* hA = (_Float16*)(ws + WS_HA_OFF);
  _Float16* hB = (_Float16*)(ws + WS_HB_OFF);

  // Zero barrier counter and h_{-1} (both buffers); capture-safe stream op.
  hipMemsetAsync(d_ws, 0, WS_INIT, stream);

  esn_persistent_kernel<<<NWG, BLOCK, 0, stream>>>(x, Win, W, out, hA, hB, bar);
}
